// MultiHeadAttention_5617817223839
// MI455X (gfx1250) — matrix operand order repaired and checked by reading, not compiled, not run on hardware
//
#include <hip/hip_runtime.h>

// Problem constants (from reference): B=4, S=2048, E=1024, H=16, DK=64
#define B_SZ   4
#define S_LEN  2048
#define EMB    1024
#define HEADS  16
#define DKH    64
#define MROWS  (B_SZ * S_LEN)   // 8192 total token rows

typedef __bf16 bf16;
typedef __attribute__((ext_vector_type(16))) __bf16 v16bf;
typedef __attribute__((ext_vector_type(8)))  __bf16 v8bf;
typedef __attribute__((ext_vector_type(4)))  __bf16 v4bf;
typedef __attribute__((ext_vector_type(8)))  float  v8f;
typedef __attribute__((ext_vector_type(4)))  unsigned int u32x4;
typedef __attribute__((ext_vector_type(8)))  int          i32x8;
typedef __attribute__((ext_vector_type(4)))  int          i32x4;

#if __has_builtin(__builtin_amdgcn_tensor_load_to_lds)
#define HAVE_TDM 1
#else
#define HAVE_TDM 0
#endif

// ---------------------------------------------------------------------------
// Fragment helpers (wave32; contiguous K order chosen by this code, same for A and B)
//  A (16x32, MxK): lane holds row m=(lane&15); 16 contiguous K at 16*hsel
//                  (hsel = lane>>4)
//  B (32x16, KxN): lane holds col n=(lane&15); 16 contiguous K at 16*hsel
// ---------------------------------------------------------------------------
static __device__ __forceinline__ v16bf make_v16(v8bf lo, v8bf hi) {
  union { v16bf v; v8bf h[2]; } u;
  u.h[0] = lo; u.h[1] = hi;
  return u.v;
}

static __device__ __forceinline__ v16bf load_a_frag(const bf16* rowptr, int hsel) {
  v8bf lo = *(const v8bf*)(rowptr + 16 * hsel);
  v8bf hi = *(const v8bf*)(rowptr + 16 * hsel + 8);
  return make_v16(lo, hi);
}

static __device__ __forceinline__ v16bf load_b_frag(const bf16* p) {
  v8bf lo = *(const v8bf*)(p);
  v8bf hi = *(const v8bf*)(p + 8);
  return make_v16(lo, hi);
}

static __device__ __forceinline__ v8f wmma_bf16(v16bf a, v16bf b, v8f c) {
  // (neg_a, A, neg_b, B, c_mod, C, reuse_a, reuse_b)
  return __builtin_amdgcn_wmma_f32_16x16x32_bf16(false, a, false, b,
                                                 (short)0, c, false, false);
}

#if HAVE_TDM
// Issue one TDM 2D tile load: 64 rows x 64 halves (bf16), source row stride
// src_stride elements, dest rows padded to 72 halves (pad 4 DW per 32 DW).
static __device__ __forceinline__ void tdm_load_tile64(const bf16* gsrc,
                                                       unsigned lds_bytes,
                                                       int src_stride) {
  unsigned long long ga = (unsigned long long)(const void*)gsrc;
  u32x4 g0;
  g0[0] = 1u;                                   // count=1 valid descriptor
  g0[1] = lds_bytes;                            // lds_addr
  g0[2] = (unsigned)ga;                         // global_addr lo
  g0[3] = ((unsigned)(ga >> 32) & 0x01FFFFFFu) | 0x80000000u;  // hi | type=2
  i32x8 g1;
  g1[0] = (int)((1u << 16) | (1u << 20) | (4u << 22) | (3u << 25));
  //        data_size=2B    pad_enable   pad_int=32DW  pad_amt=4DW
  g1[1] = (int)(64u << 16);   // tensor_dim0 lo16 = 64
  g1[2] = (int)(64u << 16);   // tensor_dim1 lo16 = 64
  g1[3] = (int)(64u << 16);   // tile_dim0 = 64
  g1[4] = 64;                 // tile_dim1 = 64
  g1[5] = src_stride;         // tensor_dim0_stride lo32
  g1[6] = 0;
  g1[7] = 0;
  i32x4 z4 = {0, 0, 0, 0};
#if defined(__clang_major__) && (__clang_major__ >= 23)
  i32x8 z8 = {0, 0, 0, 0, 0, 0, 0, 0};
  __builtin_amdgcn_tensor_load_to_lds(g0, g1, z4, z4, z8, 0);
#else
  __builtin_amdgcn_tensor_load_to_lds(g0, g1, z4, z4, 0);
#endif
}
#endif

// ---------------------------------------------------------------------------
// f32 -> bf16 conversion (vectorized 4/thread)
// ---------------------------------------------------------------------------
__global__ __launch_bounds__(256)
void cvt_f32_bf16(const float* __restrict__ x, bf16* __restrict__ y, int n) {
  int i = (blockIdx.x * blockDim.x + threadIdx.x) * 4;
  if (i >= n) return;
  float4 v = *(const float4*)(x + i);
  v4bf o;
  o[0] = (bf16)v.x; o[1] = (bf16)v.y; o[2] = (bf16)v.z; o[3] = (bf16)v.w;
  *(v4bf*)(y + i) = o;
}

// ---------------------------------------------------------------------------
// WMMA GEMM:  C[M=8192, N=1024] = A[M,K=1024] * W[N,K]^T + bias
//   mode 0: store bf16 row-major [M,N]
//   mode 1: store f32 row-major  [M,N]   (final output projection)
//   mode 2: store bf16 transposed as Vt[b, h, d, s]  (for attention PV)
//
// Block = 256 threads = 8 waves. Wave tile = 32(M) x 64(N); block = 256 x 64.
// Block-shared W tile (64 rows x 32 k) staged into LDS by the Tensor Data
// Mover, double buffered, pad feature -> 40-half LDS rows (conflict-free).
// ---------------------------------------------------------------------------
#define WROW_LDS 40                       // halves per LDS row (32 data + 8 pad)
#define WTILE_HALVES (64 * WROW_LDS)      // 2560 halves = 5120 B per buffer

__global__ __launch_bounds__(256)
void gemm_wmma(const bf16* __restrict__ A, const bf16* __restrict__ W,
               const float* __restrict__ bias, float* __restrict__ Cf,
               bf16* __restrict__ Cb, int mode) {
  __shared__ __align__(16) bf16 wtile[2][WTILE_HALVES];

  const int lane = threadIdx.x & 31;
  const int wave = threadIdx.x >> 5;
  const int hsel = lane >> 4;
  const int nl   = lane & 15;
  const int m0 = blockIdx.x * 256 + wave * 32;
  const int n0 = blockIdx.y * 64;

#if HAVE_TDM
  const unsigned ldsbase = (unsigned)(unsigned long long)(void*)&wtile[0][0];
#endif

  auto stage = [&](int buf, int k0) {
#if HAVE_TDM
    if (wave == 0) {
      unsigned long long ga =
          (unsigned long long)(const void*)(W + (size_t)n0 * EMB + k0);
      u32x4 g0;
      g0[0] = 1u;
      g0[1] = ldsbase + (unsigned)buf * (WTILE_HALVES * 2);
      g0[2] = (unsigned)ga;
      g0[3] = ((unsigned)(ga >> 32) & 0x01FFFFFFu) | 0x80000000u;
      i32x8 g1;
      g1[0] = (int)((1u << 16) | (1u << 20) | (3u << 22) | (3u << 25));
      //        data_size=2B    pad_enable   pad_int=16DW  pad_amt=4DW
      g1[1] = (int)(32u << 16);   // tensor_dim0 = 32
      g1[2] = (int)(64u << 16);   // tensor_dim1 = 64
      g1[3] = (int)(32u << 16);   // tile_dim0 = 32
      g1[4] = 64;                 // tile_dim1 = 64
      g1[5] = 1024;               // tensor_dim0_stride
      g1[6] = 0;
      g1[7] = 0;
      i32x4 z4 = {0, 0, 0, 0};
#if defined(__clang_major__) && (__clang_major__ >= 23)
      i32x8 z8 = {0, 0, 0, 0, 0, 0, 0, 0};
      __builtin_amdgcn_tensor_load_to_lds(g0, g1, z4, z4, z8, 0);
#else
      __builtin_amdgcn_tensor_load_to_lds(g0, g1, z4, z4, 0);
#endif
    }
#else
    const int row = threadIdx.x >> 2;
    const int c   = threadIdx.x & 3;
    *(v8bf*)&wtile[buf][row * WROW_LDS + c * 8] =
        *(const v8bf*)(W + (size_t)(n0 + row) * EMB + k0 + c * 8);
#endif
  };

  auto settle = [&]() {
#if HAVE_TDM
    if (wave == 0) __builtin_amdgcn_s_wait_tensorcnt(0);
#endif
    __syncthreads();
  };

  const bf16* arow0 = A + (size_t)(m0 + nl) * EMB;
  const bf16* arow1 = A + (size_t)(m0 + 16 + nl) * EMB;

  v8f acc[8] = {};   // [0..3] = rows m0..m0+15, [4..7] = rows m0+16..m0+31

  stage(0, 0);
  settle();

  int buf = 0;
  for (int k0 = 0; k0 < EMB; k0 += 32) {
    if (k0 + 32 < EMB) stage(buf ^ 1, k0 + 32);   // TDM prefetch next k-chunk

    v16bf a0 = load_a_frag(arow0 + k0, hsel);
    v16bf a1 = load_a_frag(arow1 + k0, hsel);
    const bf16* wb = &wtile[buf][0];
#pragma unroll
    for (int t = 0; t < 4; t++) {
      v16bf b = load_b_frag(wb + (16 * t + nl) * WROW_LDS + 16 * hsel);
      acc[t]     = wmma_bf16(a0, b, acc[t]);
      acc[4 + t] = wmma_bf16(a1, b, acc[4 + t]);
    }

    settle();   // next buffer staged; all waves done reading current buffer
    buf ^= 1;
  }

#pragma unroll
  for (int mi = 0; mi < 2; mi++) {
#pragma unroll
    for (int t = 0; t < 4; t++) {
      const int n = n0 + 16 * t + nl;
      const float bv = bias[n];
#pragma unroll
      for (int r = 0; r < 8; r++) {
        const int m = m0 + 16 * mi + r + 8 * hsel;
        const float v = acc[4 * mi + t][r] + bv;
        if (mode == 1) {
          Cf[(size_t)m * EMB + n] = v;
        } else if (mode == 0) {
          Cb[(size_t)m * EMB + n] = (bf16)v;
        } else {
          // V projection stored transposed: Vt[(b*H + h)*DK + d][s]
          const int bb = m >> 11;
          const int s  = m & (S_LEN - 1);
          const int h  = n >> 6;
          const int d  = n & (DKH - 1);
          Cb[(((size_t)(bb * HEADS + h)) * DKH + d) * S_LEN + s] = (bf16)v;
        }
      }
    }
  }
}

// ---------------------------------------------------------------------------
// Flash attention, bf16 WMMA, online softmax.
// Grid: (S/128, H, B); block 256 = 8 waves; wave owns 16 query rows.
// Shared K tile (64kv x 64d) and V tile (64d x 64kv, from pre-transposed Vt)
// staged into double-buffered LDS by TWO TDM descriptors per kv step: wave 0
// issues K, wave 1 issues V, each waits its own TENSORcnt; the end-of-step
// barrier publishes both tiles. 8x reduction in global K/V traffic.
// kv tiles of 64; per tile: 8 WMMAs (QK^T) + LDS relayout + 8 WMMAs (PV).
// ---------------------------------------------------------------------------
#define KV_STRIDE 72                     // halves per LDS row (64 data + 8 pad)
#define KVTILE_HALVES (64 * KV_STRIDE)   // 4608 halves = 9216 B per buffer

__global__ __launch_bounds__(256)
void flash_attn(const bf16* __restrict__ Qm, const bf16* __restrict__ Km,
                const bf16* __restrict__ Vt, bf16* __restrict__ ctx) {
  __shared__ __align__(16) bf16 ktile[2][KVTILE_HALVES];
  __shared__ __align__(16) bf16 vtile[2][KVTILE_HALVES];
  __shared__ __align__(16) bf16 plds[8][16][72];   // per-wave P tile, padded

  const int lane = threadIdx.x & 31;
  const int wave = threadIdx.x >> 5;
  const int hsel = lane >> 4;
  const int nl   = lane & 15;
  const int b = blockIdx.z, h = blockIdx.y;
  const int q0 = blockIdx.x * 128 + wave * 16;

  const bf16* Qp = Qm + ((size_t)b * S_LEN + q0 + nl) * EMB + h * DKH;
  const bf16* Kp = Km + ((size_t)b * S_LEN) * EMB + h * DKH;
  const bf16* Vp = Vt + ((size_t)(b * HEADS + h)) * DKH * S_LEN;

#if HAVE_TDM
  const unsigned kbase = (unsigned)(unsigned long long)(void*)&ktile[0][0];
  const unsigned vbase = (unsigned)(unsigned long long)(void*)&vtile[0][0];
#endif

  auto stage = [&](int buf, int kv0) {
#if HAVE_TDM
    if (wave == 0)
      tdm_load_tile64(Kp + (size_t)kv0 * EMB, kbase + buf * (KVTILE_HALVES * 2),
                      EMB);
    else if (wave == 1)
      tdm_load_tile64(Vp + kv0, vbase + buf * (KVTILE_HALVES * 2), S_LEN);
#else
    for (int it = threadIdx.x; it < 512; it += 256) {
      const int row = it >> 3, cc = it & 7;
      *(v8bf*)&ktile[buf][row * KV_STRIDE + cc * 8] =
          *(const v8bf*)(Kp + (size_t)(kv0 + row) * EMB + cc * 8);
      *(v8bf*)&vtile[buf][row * KV_STRIDE + cc * 8] =
          *(const v8bf*)(Vp + (size_t)row * S_LEN + kv0 + cc * 8);
    }
#endif
  };

  auto settle = [&]() {
#if HAVE_TDM
    if (wave < 2) __builtin_amdgcn_s_wait_tensorcnt(0);
#endif
    __syncthreads();
  };

  // Q A-fragments (16 rows x 64 dims = two K=32 chunks), loaded once.
  v16bf aq[2];
#pragma unroll
  for (int kc = 0; kc < 2; kc++) aq[kc] = load_a_frag(Qp + 32 * kc, hsel);

  float mrow[8], lrow[8];
  v8f O[4] = {};
#pragma unroll
  for (int r = 0; r < 8; r++) { mrow[r] = -3.0e38f; lrow[r] = 0.0f; }

  // softmax(s/sqrt(64)) via exp2: c = (1/8) * log2(e)
  const float c = 0.18033688011112042f;

  stage(0, 0);
  settle();

  int buf = 0;
  for (int kv0 = 0; kv0 < S_LEN; kv0 += 64) {
    if (kv0 + 64 < S_LEN) stage(buf ^ 1, kv0 + 64);  // TDM prefetch next tiles

    // ---- S = Q * K^T over this 16x64 tile (K from LDS) ----
    v8f Sa[4] = {};
#pragma unroll
    for (int t = 0; t < 4; t++) {
      const bf16* kr = &ktile[buf][(16 * t + nl) * KV_STRIDE + 16 * hsel];
#pragma unroll
      for (int kc = 0; kc < 2; kc++) {
        v16bf bk = load_b_frag(kr + 32 * kc);
        Sa[t] = wmma_bf16(aq[kc], bk, Sa[t]);
      }
    }

    // ---- online softmax: row max (half-wave reduction), rescale O and l ----
    float psum[8];
#pragma unroll
    for (int r = 0; r < 8; r++) {
      float v = Sa[0][r];
      v = fmaxf(v, Sa[1][r]); v = fmaxf(v, Sa[2][r]); v = fmaxf(v, Sa[3][r]);
      v = fmaxf(v, __shfl_xor(v, 1, 16));
      v = fmaxf(v, __shfl_xor(v, 2, 16));
      v = fmaxf(v, __shfl_xor(v, 4, 16));
      v = fmaxf(v, __shfl_xor(v, 8, 16));
      const float mnew = fmaxf(mrow[r], v);
      const float corr = __builtin_amdgcn_exp2f((mrow[r] - mnew) * c);
      mrow[r] = mnew;
      lrow[r] *= corr;
#pragma unroll
      for (int t = 0; t < 4; t++) O[t][r] *= corr;
      psum[r] = 0.0f;
    }

    // ---- P = exp2(c*(s-m)); stash in LDS (C-layout -> A-layout bridge) ----
#pragma unroll
    for (int t = 0; t < 4; t++) {
#pragma unroll
      for (int r = 0; r < 8; r++) {
        const float p = __builtin_amdgcn_exp2f((Sa[t][r] - mrow[r]) * c);
        psum[r] += p;
        plds[wave][r + 8 * hsel][16 * t + nl] = (bf16)p;
      }
    }
#pragma unroll
    for (int r = 0; r < 8; r++) {
      float s = psum[r];
      s += __shfl_xor(s, 1, 16);
      s += __shfl_xor(s, 2, 16);
      s += __shfl_xor(s, 4, 16);
      s += __shfl_xor(s, 8, 16);
      lrow[r] += s;
    }
    __syncthreads();   // publish P (cross-lane LDS RAW within the block)

    // ---- O += P * V  (P from LDS as A-frags, V tile from LDS as B-frags) ----
    v16bf ap[2];
#pragma unroll
    for (int kc = 0; kc < 2; kc++) {
      const bf16* pr = &plds[wave][nl][32 * kc + 16 * hsel];
      v8bf lo = *(const v8bf*)pr;
      v8bf hi = *(const v8bf*)(pr + 8);
      ap[kc] = make_v16(lo, hi);
    }
#pragma unroll
    for (int t = 0; t < 4; t++) {
      const bf16* vr = &vtile[buf][(16 * t + nl) * KV_STRIDE + 16 * hsel];
#pragma unroll
      for (int kc = 0; kc < 2; kc++) {
        v16bf bv = load_b_frag(vr + 32 * kc);
        O[t] = wmma_bf16(ap[kc], bv, O[t]);
      }
    }

    settle();   // next tiles staged; all waves done reading current buffers
    buf ^= 1;
  }

  // ---- epilogue: O / l, store bf16 ctx [B,S,E] ----
  float inv[8];
#pragma unroll
  for (int r = 0; r < 8; r++) inv[r] = 1.0f / lrow[r];
#pragma unroll
  for (int t = 0; t < 4; t++) {
#pragma unroll
    for (int r = 0; r < 8; r++) {
      const float o = O[t][r] * inv[r];
      ctx[((size_t)b * S_LEN + q0 + r + 8 * hsel) * EMB + h * DKH + 16 * t + nl] =
          (bf16)o;
    }
  }
}

// ---------------------------------------------------------------------------
// Host-side orchestration
// ---------------------------------------------------------------------------
extern "C" void kernel_launch(void* const* d_in, const int* in_sizes, int n_in,
                              void* d_out, int out_size, void* d_ws, size_t ws_size,
                              hipStream_t stream) {
  (void)in_sizes; (void)n_in; (void)out_size; (void)ws_size;

  const float* query = (const float*)d_in[0];
  const float* key_  = (const float*)d_in[1];
  const float* value = (const float*)d_in[2];
  const float* Wq = (const float*)d_in[3];  const float* bq = (const float*)d_in[4];
  const float* Wk = (const float*)d_in[5];  const float* bk = (const float*)d_in[6];
  const float* Wv = (const float*)d_in[7];  const float* bv = (const float*)d_in[8];
  const float* Wo = (const float*)d_in[9];  const float* bo = (const float*)d_in[10];
  float* out = (float*)d_out;

  // Workspace carve-out (all 256B aligned)
  char* ws = (char*)d_ws;
  size_t off = 0;
  auto carve = [&](size_t bytes) -> void* {
    void* p = ws + off;
    off += (bytes + 255) & ~(size_t)255;
    return p;
  };
  const size_t actB = (size_t)MROWS * EMB * sizeof(bf16);  // 16 MiB
  const size_t wB   = (size_t)EMB * EMB * sizeof(bf16);    //  2 MiB
  bf16* xq  = (bf16*)carve(actB);
  bf16* xk  = (bf16*)carve(actB);
  bf16* xv  = (bf16*)carve(actB);
  bf16* wqb = (bf16*)carve(wB);
  bf16* wkb = (bf16*)carve(wB);
  bf16* wvb = (bf16*)carve(wB);
  bf16* wob = (bf16*)carve(wB);
  bf16* Qb  = (bf16*)carve(actB);
  bf16* Kb  = (bf16*)carve(actB);
  bf16* Vt  = (bf16*)carve(actB);   // transposed [B,H,DK,S]
  bf16* ctx = (bf16*)carve(actB);
  // total ~120 MiB

  const int nAct = MROWS * EMB;     // 8,388,608
  const int nW   = EMB * EMB;       // 1,048,576
  cvt_f32_bf16<<<nAct / (256 * 4), 256, 0, stream>>>(query, xq, nAct);
  cvt_f32_bf16<<<nAct / (256 * 4), 256, 0, stream>>>(key_,  xk, nAct);
  cvt_f32_bf16<<<nAct / (256 * 4), 256, 0, stream>>>(value, xv, nAct);
  cvt_f32_bf16<<<nW   / (256 * 4), 256, 0, stream>>>(Wq, wqb, nW);
  cvt_f32_bf16<<<nW   / (256 * 4), 256, 0, stream>>>(Wk, wkb, nW);
  cvt_f32_bf16<<<nW   / (256 * 4), 256, 0, stream>>>(Wv, wvb, nW);
  cvt_f32_bf16<<<nW   / (256 * 4), 256, 0, stream>>>(Wo, wob, nW);

  dim3 gemmGrid(MROWS / 256, EMB / 64, 1);        // 32 x 16 blocks
  gemm_wmma<<<gemmGrid, 256, 0, stream>>>(xq, wqb, bq, nullptr, Qb, 0);
  gemm_wmma<<<gemmGrid, 256, 0, stream>>>(xk, wkb, bk, nullptr, Kb, 0);
  gemm_wmma<<<gemmGrid, 256, 0, stream>>>(xv, wvb, bv, nullptr, Vt, 2);

  dim3 attnGrid(S_LEN / 128, HEADS, B_SZ);        // 16 x 16 x 4 blocks
  flash_attn<<<attnGrid, 256, 0, stream>>>(Qb, Kb, Vt, ctx);

  gemm_wmma<<<gemmGrid, 256, 0, stream>>>(ctx, wob, bo, out, nullptr, 1);
}
